// VCSA_1906965479601
// MI455X (gfx1250) — matrix operand order repaired and checked by reading, not compiled, not run on hardware
//
#include <hip/hip_runtime.h>
#include <hip/hip_bf16.h>

typedef __attribute__((ext_vector_type(16))) _Float16 v16h;
typedef __attribute__((ext_vector_type(8)))  _Float16 v8h;
typedef __attribute__((ext_vector_type(4)))  _Float16 v4h;
typedef __attribute__((ext_vector_type(8)))  float    v8f;

#define NODE_DIM 128
#define HIDDEN   64
#define WAVES_PER_BLOCK 8

// ---------------------------------------------------------------------------
// Kernel 1: convert x (N x 128, fp32) -> f16 in workspace. x is re-read ~12.5x
// by the edge gather, so converting once halves gather bytes and lets the
// gather feed v_wmma_f32_16x16x32_f16 directly. 25.6MB f16 sits in L2 (192MB).
// ---------------------------------------------------------------------------
__global__ void cvt_x_f16(const float* __restrict__ x,
                          _Float16* __restrict__ xh, int n4) {
  int i = blockIdx.x * blockDim.x + threadIdx.x;
  if (i < n4) {
    float4 v = ((const float4*)x)[i];
    v4h h;
    h.x = (_Float16)v.x; h.y = (_Float16)v.y;
    h.z = (_Float16)v.z; h.w = (_Float16)v.w;
    ((v4h*)xh)[i] = h;
  }
}

// ---------------------------------------------------------------------------
// Kernel 2: edge MLP. One wave computes TWO tiles of 16 edges (32 edges).
//   GEMM1: edge_attr(16x256) @ W1(256x64); per (kb,nb) B-fragment from LDS
//          feeds two independent WMMAs (tile0/tile1) -> ds:wmma ratio 1:1.
//   Layer2: relu(+b1) dot W2, 16-lane butterfly reduce, sigmoid(+b2), store.
// W1 staged once per block into LDS in WMMA-B fragment layout.
// ---------------------------------------------------------------------------
__global__ void __launch_bounds__(256)
edge_mlp(const _Float16* __restrict__ xh,
         const int* __restrict__ eidx,       // [2*E] row then col
         const float* __restrict__ W1,       // [256*64] row-major (K,N)
         const float* __restrict__ b1,       // [64]
         const float* __restrict__ W2,       // [64]
         const float* __restrict__ b2,       // [1]
         float* __restrict__ out,            // [E]
         int E_, int nPairs) {
  // 8 kb * 4 nb * 32 lanes * 16 halves = 32KB
  __shared__ __align__(32) _Float16 w1f[8 * 4 * 32 * 16];

  const int t = threadIdx.x;
  // Convert W1 fp32 -> f16 fragments in LDS (whole block cooperates).
  for (int i = t; i < 256 * 64; i += 256) {
    int k = i >> 6;          // row (K)
    int n = i & 63;          // col (N)
    int kb   = k >> 5;
    int kin  = k & 31;
    int half = (kin >> 3) & 1; // 0: lanes 0-15 (K 0-7,16-23), 1: lanes 16-31 (K 8-15,24-31)
    int tt   = (kin & 7) | ((kin >> 4) << 3);
    int nb   = n >> 4;
    int ln   = n & 15;
    int lane = half * 16 + ln;
    w1f[((kb * 4 + nb) * 32 + lane) * 16 + tt] = (_Float16)W1[i];
  }
  __syncthreads();

  const int wave = t >> 5;
  const int lid  = t & 31;
  const int pair = blockIdx.x * WAVES_PER_BLOCK + wave;
  if (pair >= nPairs) return;            // wave-uniform: EXEC stays all-1s for WMMA

  const int base0 = (pair * 2) * 16;
  const int base1 = base0 + 16;
  const int m16 = lid & 15;              // edge-in-tile this lane gathers for
  const bool hi = lid >= 16;

  int eid0 = base0 + m16; if (eid0 >= E_) eid0 = E_ - 1;  // clamp; stores predicated
  int eid1 = base1 + m16; if (eid1 >= E_) eid1 = E_ - 1;
  const int r0 = eidx[eid0];
  const int c0 = eidx[E_ + eid0];
  const int r1 = eidx[eid1];
  const int c1 = eidx[E_ + eid1];

  // Per-lane slice of b1 / W2 (column n = nb*16 + m16 in C-fragment layout).
  float b1v[4], w2v[4];
#pragma unroll
  for (int nb = 0; nb < 4; ++nb) {
    b1v[nb] = b1[nb * 16 + m16];
    w2v[nb] = W2[nb * 16 + m16];
  }
  const float bias2 = b2[0];

  v8f acc0[4] = {};                      // tile0: 4 n-tiles of 16x64 h
  v8f acc1[4] = {};                      // tile1

  const int laneoff = hi ? 8 : 0;

#pragma unroll
  for (int kb = 0; kb < 8; ++kb) {
    // First 128 features from row node, next 128 from col node.
    const int node0 = (kb < 4) ? r0 : c0;
    const int node1 = (kb < 4) ? r1 : c1;
    const int fb    = (kb & 3) * 32 + laneoff;
    // A layout (16-bit 16x32): lanes 0-15 hold K fb..fb+7 and fb+16..fb+23,
    // lanes 16-31 hold K fb+8.. and fb+24.. — two 16B loads each.
    const _Float16* p0 = xh + (size_t)node0 * NODE_DIM + fb;
    const _Float16* p1 = xh + (size_t)node1 * NODE_DIM + fb;
    v8h a0lo = *(const v8h*)p0;
    v8h a0hi = *(const v8h*)(p0 + 16);
    v8h a1lo = *(const v8h*)p1;
    v8h a1hi = *(const v8h*)(p1 + 16);
    v16h a0 = __builtin_shufflevector(a0lo, a0hi,
                                      0, 1, 2, 3, 4, 5, 6, 7,
                                      8, 9, 10, 11, 12, 13, 14, 15);
    v16h a1 = __builtin_shufflevector(a1lo, a1hi,
                                      0, 1, 2, 3, 4, 5, 6, 7,
                                      8, 9, 10, 11, 12, 13, 14, 15);
#pragma unroll
    for (int nb = 0; nb < 4; ++nb) {
      v16h bfrag = *(const v16h*)&w1f[((kb * 4 + nb) * 32 + lid) * 16];
      acc0[nb] = __builtin_amdgcn_wmma_f32_16x16x32_f16(
          false, a0, false, bfrag, (short)0, acc0[nb], false, false);
      acc1[nb] = __builtin_amdgcn_wmma_f32_16x16x32_f16(
          false, a1, false, bfrag, (short)0, acc1[nb], false, false);
    }
  }

  // ---- Layer 2 epilogue, tile0 then tile1 (limits live registers) ----
  // C layout: VGPR j holds M=j (lanes 0-15) / M=j+8 (lanes 16-31), N = lane&15.
#pragma unroll
  for (int which = 0; which < 2; ++which) {
    const v8f* acc   = which ? acc1 : acc0;
    const int  tbase = which ? base1 : base0;
    float partial[8];
#pragma unroll
    for (int j = 0; j < 8; ++j) partial[j] = 0.0f;
#pragma unroll
    for (int nb = 0; nb < 4; ++nb) {
#pragma unroll
      for (int j = 0; j < 8; ++j) {
        float v = acc[nb][j] + b1v[nb];
        v = v > 0.0f ? v : 0.0f;         // relu
        partial[j] = fmaf(v, w2v[nb], partial[j]);
      }
    }
    // Butterfly reduce over the 16 lanes of each half (N dimension).
#pragma unroll
    for (int j = 0; j < 8; ++j) {
      float s = partial[j];
      s += __shfl_xor(s, 1, 32);
      s += __shfl_xor(s, 2, 32);
      s += __shfl_xor(s, 4, 32);
      s += __shfl_xor(s, 8, 32);
      const int m  = j + (hi ? 8 : 0);
      const int oe = tbase + m;
      if (m16 == j && oe < E_) {
        const float z = s + bias2;
        out[oe] = 1.0f / (1.0f + __expf(-z));   // sigmoid
      }
    }
  }
}

// ---------------------------------------------------------------------------
extern "C" void kernel_launch(void* const* d_in, const int* in_sizes, int n_in,
                              void* d_out, int out_size, void* d_ws, size_t ws_size,
                              hipStream_t stream) {
  const float* x    = (const float*)d_in[0];
  const int*   eidx = (const int*)d_in[1];   // [2*E], row then col
  const float* W1   = (const float*)d_in[2];
  const float* b1   = (const float*)d_in[3];
  const float* W2   = (const float*)d_in[4];
  const float* b2   = (const float*)d_in[5];
  float* out = (float*)d_out;

  const int E_ = out_size;                   // 625000
  const int NX = in_sizes[0];                // N * 128 = 12.8M

  _Float16* xh = (_Float16*)d_ws;            // 25.6 MB f16 copy of x

  const int n4 = NX / 4;
  cvt_x_f16<<<(n4 + 255) / 256, 256, 0, stream>>>(x, xh, n4);

  const int nTiles = (E_ + 15) / 16;
  const int nPairs = (nTiles + 1) / 2;
  const int blocks = (nPairs + WAVES_PER_BLOCK - 1) / WAVES_PER_BLOCK;
  edge_mlp<<<blocks, 256, 0, stream>>>(xh, eidx, W1, b1, W2, b2, out, E_, nPairs);
}
